// Encoder_27496380629558
// MI455X (gfx1250) — matrix operand order repaired and checked by reading, not compiled, not run on hardware
//
#include <hip/hip_runtime.h>
#include <cstdint>

typedef __attribute__((ext_vector_type(16))) _Float16 v16h;
typedef __attribute__((ext_vector_type(8)))  float    v8f;

// ---------------------------------------------------------------------------
// Problem constants
// ---------------------------------------------------------------------------
// B=64, V = {16384,4096,1024,256,64}, C = {3,32,64,128,256}, S=9, LATENT=256

// ---------------------------------------------------------------------------
// Workspace layout (bytes). All offsets 256B-aligned.
// ---------------------------------------------------------------------------
static constexpr size_t SZ_Y   = 64ull * 16384 * 32 * 4;      // 134,217,728 (max y: block 0)
static constexpr size_t SZ_H1  = 64ull * 4096 * 32 * 4;       //  33,554,432
static constexpr size_t SZ_H2  = 64ull * 1024 * 64 * 4;       //  16,777,216
static constexpr size_t SZ_H3  = 64ull * 256 * 128 * 4;       //   8,388,608
static constexpr size_t SZ_H4  = 64ull * 64 * 256 * 4;        //   4,194,304
static constexpr size_t SZ_BP0 = 1ull  * 2  * 32 * 16 * 2;    //       2,048
static constexpr size_t SZ_BP1 = 9ull  * 4  * 32 * 16 * 2;    //      36,864
static constexpr size_t SZ_BP2 = 18ull * 8  * 32 * 16 * 2;    //     147,456
static constexpr size_t SZ_BP3 = 36ull * 16 * 32 * 16 * 2;    //     589,824
static constexpr size_t SZ_BPF = 512ull* 16 * 32 * 16 * 2;    //   8,388,608
static constexpr size_t SZ_PART= 16ull * 64 * 256 * 4;        //   1,048,576

static constexpr size_t OFF_Y   = 0;
static constexpr size_t OFF_H1  = OFF_Y   + SZ_Y;
static constexpr size_t OFF_H2  = OFF_H1  + SZ_H1;
static constexpr size_t OFF_H3  = OFF_H2  + SZ_H2;
static constexpr size_t OFF_H4  = OFF_H3  + SZ_H3;
static constexpr size_t OFF_BP0 = OFF_H4  + SZ_H4;
static constexpr size_t OFF_BP1 = OFF_BP0 + SZ_BP0;
static constexpr size_t OFF_BP2 = OFF_BP1 + SZ_BP1;
static constexpr size_t OFF_BP3 = OFF_BP2 + SZ_BP2;
static constexpr size_t OFF_BPF = OFF_BP3 + SZ_BP3;
static constexpr size_t OFF_PART= OFF_BPF + SZ_BPF;
// total ~207.4 MB

// ---------------------------------------------------------------------------
// Pack a weight matrix W[K,N] (f32, row-major) into WMMA B-fragment order f16.
// Fragment index: ((kb*NT + nt)*32 + lane)*16 + e
//   n = nt*16 + (lane & 15),  k = kb*32 + (e>>3)*16 + (lane>>4)*8 + (e&7)   (zero-pad k>=K)
// ---------------------------------------------------------------------------
__global__ void pack_w_kernel(const float* __restrict__ W, _Float16* __restrict__ Bp,
                              int K, int N, int KB, int NT) {
    int tid = blockIdx.x * blockDim.x + threadIdx.x;
    int total = KB * NT * 32;
    if (tid >= total) return;
    int l  = tid & 31;
    int t  = tid >> 5;
    int nt = t % NT;
    int kb = t / NT;
    int n  = nt * 16 + (l & 15);
    v16h frag;
#pragma unroll
    for (int e = 0; e < 16; ++e) {
        int k = kb * 32 + ((e >> 3) << 4) + ((l >> 4) << 3) + (e & 7);
        frag[e] = (k < K) ? (_Float16)W[(size_t)k * N + n] : (_Float16)0.0f;
    }
    *((v16h*)Bp + tid) = frag;
}

// ---------------------------------------------------------------------------
// Fused spiral-gather + GEMM + bias + ELU.
// A[m,k] = X[b, spiral[v][k/C_IN], k%C_IN], m = b*V + v, K = 9*C_IN.
// One wave computes a 16x16 tile of Y[M,N]; accumulate via wmma f32.16x16x32.f16.
// For pow2 C_IN>=32 a whole K-step per lane has a single neighbor s and two
// contiguous 8-float runs -> 4x float4 loads. C_IN==3 (block 0) scalar path.
// ---------------------------------------------------------------------------
template<int C_IN>
__global__ void spiral_gemm_kernel(const float* __restrict__ X, const int* __restrict__ spiral,
                                   const _Float16* __restrict__ Bp, const float* __restrict__ bias,
                                   float* __restrict__ Y,
                                   int vShift, int Kreal, int KB, int N, int NT, int numTiles) {
    int wave = (blockIdx.x * blockDim.x + threadIdx.x) >> 5;
    int l = threadIdx.x & 31;
    if (wave >= numTiles) return;
    int nt = wave % NT;
    int mt = wave / NT;

    int m = mt * 16 + (l & 15);
    int b = m >> vShift;
    int v = m & ((1 << vShift) - 1);
    const size_t rowBase = ((size_t)b << vShift) * C_IN;   // b*V*C_IN
    const int* srow = spiral + v * 9;
    const int hi8 = (l >> 4) << 3;

    v8f acc = {0.f, 0.f, 0.f, 0.f, 0.f, 0.f, 0.f, 0.f};

    for (int kb = 0; kb < KB; ++kb) {
        v16h a;
        if constexpr (C_IN >= 32) {
            constexpr int LOG2C = (C_IN == 32) ? 5 : (C_IN == 64) ? 6 : 7;
            int kbase = kb << 5;
            int s     = kbase >> LOG2C;
            int cbase = kbase & (C_IN - 1);
            int vert  = srow[s];
            const float* xr = X + rowBase + (size_t)vert * C_IN + cbase + hi8;
            float4 f0 = *reinterpret_cast<const float4*>(xr);
            float4 f1 = *reinterpret_cast<const float4*>(xr + 4);
            float4 f2 = *reinterpret_cast<const float4*>(xr + 16);
            float4 f3 = *reinterpret_cast<const float4*>(xr + 20);
            a[0]=(_Float16)f0.x;  a[1]=(_Float16)f0.y;  a[2]=(_Float16)f0.z;  a[3]=(_Float16)f0.w;
            a[4]=(_Float16)f1.x;  a[5]=(_Float16)f1.y;  a[6]=(_Float16)f1.z;  a[7]=(_Float16)f1.w;
            a[8]=(_Float16)f2.x;  a[9]=(_Float16)f2.y;  a[10]=(_Float16)f2.z; a[11]=(_Float16)f2.w;
            a[12]=(_Float16)f3.x; a[13]=(_Float16)f3.y; a[14]=(_Float16)f3.z; a[15]=(_Float16)f3.w;
        } else {
#pragma unroll
            for (int e = 0; e < 16; ++e) {
                int k = (kb << 5) + ((e >> 3) << 4) + hi8 + (e & 7);
                float val = 0.0f;
                if (k < Kreal) {
                    int s = k / C_IN;
                    int c = k - s * C_IN;
                    int vert = srow[s];
                    val = X[rowBase + (size_t)vert * C_IN + c];
                }
                a[e] = (_Float16)val;
            }
        }
        v16h bfrag = *((const v16h*)Bp + (size_t)(kb * NT + nt) * 32 + l);
        acc = __builtin_amdgcn_wmma_f32_16x16x32_f16(false, a, false, bfrag,
                                                     (short)0, acc, false, false);
    }

    int n0 = nt * 16 + (l & 15);
    float bn = bias[n0];
#pragma unroll
    for (int r = 0; r < 8; ++r) {
        int mg = mt * 16 + hi8 + r;
        float t = acc[r] + bn;
        t = (t > 0.0f) ? t : (__expf(t) - 1.0f);    // ELU (alpha=1)
        Y[(size_t)mg * N + n0] = t;
    }
}

// ---------------------------------------------------------------------------
// Sparse pool: H[b,r,c] = sum_{j<3} vals[3r+j] * Y[b, cols[3r+j], c]
// C = 1<<cs, Vout = 1<<vs.
// ---------------------------------------------------------------------------
__global__ void pool_kernel(const float* __restrict__ Y, const int* __restrict__ cols,
                            const float* __restrict__ vals, float* __restrict__ H,
                            int Vin, int cs, int vs, int total) {
    int idx = blockIdx.x * blockDim.x + threadIdx.x;
    if (idx >= total) return;
    int c = idx & ((1 << cs) - 1);
    int r = (idx >> cs) & ((1 << vs) - 1);
    int b = idx >> (cs + vs);
    size_t base = ((size_t)b * Vin) << cs;
    float acc = 0.0f;
#pragma unroll
    for (int j = 0; j < 3; ++j) {
        int k = r * 3 + j;
        acc += vals[k] * Y[base + (((size_t)cols[k]) << cs) + c];
    }
    H[idx] = acc;
}

// ---------------------------------------------------------------------------
// Final linear, split-K=16: part[split][64][256] = A_chunk @ Wf_chunk.
// A = h4 viewed as [64, 16384] f32 (contiguous).
// ---------------------------------------------------------------------------
__global__ void final_gemm_kernel(const float* __restrict__ A, const _Float16* __restrict__ Bp,
                                  float* __restrict__ part) {
    const int SPLIT = 16, NT = 16, KBC = 32;          // 512 K-steps total
    int wave = (blockIdx.x * blockDim.x + threadIdx.x) >> 5;
    if (wave >= 4 * NT * SPLIT) return;
    int l = threadIdx.x & 31;
    int split = wave & (SPLIT - 1);
    int t  = wave >> 4;
    int nt = t & (NT - 1);
    int mt = t >> 4;                                   // 0..3
    int hi8 = (l >> 4) << 3;
    const float* arow = A + (size_t)(mt * 16 + (l & 15)) * 16384;

    v8f acc = {0.f, 0.f, 0.f, 0.f, 0.f, 0.f, 0.f, 0.f};
    int kb0 = split * KBC;
    for (int kb = kb0; kb < kb0 + KBC; ++kb) {
        const float* xr = arow + (kb << 5) + hi8;
        float4 f0 = *reinterpret_cast<const float4*>(xr);
        float4 f1 = *reinterpret_cast<const float4*>(xr + 4);
        float4 f2 = *reinterpret_cast<const float4*>(xr + 16);
        float4 f3 = *reinterpret_cast<const float4*>(xr + 20);
        v16h a;
        a[0]=(_Float16)f0.x;  a[1]=(_Float16)f0.y;  a[2]=(_Float16)f0.z;  a[3]=(_Float16)f0.w;
        a[4]=(_Float16)f1.x;  a[5]=(_Float16)f1.y;  a[6]=(_Float16)f1.z;  a[7]=(_Float16)f1.w;
        a[8]=(_Float16)f2.x;  a[9]=(_Float16)f2.y;  a[10]=(_Float16)f2.z; a[11]=(_Float16)f2.w;
        a[12]=(_Float16)f3.x; a[13]=(_Float16)f3.y; a[14]=(_Float16)f3.z; a[15]=(_Float16)f3.w;
        v16h bfrag = *((const v16h*)Bp + (size_t)(kb * NT + nt) * 32 + l);
        acc = __builtin_amdgcn_wmma_f32_16x16x32_f16(false, a, false, bfrag,
                                                     (short)0, acc, false, false);
    }
    int n = nt * 16 + (l & 15);
#pragma unroll
    for (int r = 0; r < 8; ++r) {
        int mg = mt * 16 + hi8 + r;
        part[(size_t)split * 16384 + (size_t)mg * 256 + n] = acc[r];
    }
}

// Deterministic fixed-order split-K reduction + bias.
__global__ void final_reduce_kernel(const float* __restrict__ part, const float* __restrict__ bf,
                                    float* __restrict__ out) {
    int idx = blockIdx.x * blockDim.x + threadIdx.x;    // 16384 = 64*256
    if (idx >= 16384) return;
    float s = bf[idx & 255];
#pragma unroll
    for (int j = 0; j < 16; ++j) s += part[(size_t)j * 16384 + idx];
    out[idx] = s;
}

// ---------------------------------------------------------------------------
// Host-side launch sequence
// ---------------------------------------------------------------------------
extern "C" void kernel_launch(void* const* d_in, const int* in_sizes, int n_in,
                              void* d_out, int out_size, void* d_ws, size_t ws_size,
                              hipStream_t stream) {
    (void)in_sizes; (void)n_in; (void)out_size; (void)ws_size;

    const float* x = (const float*)d_in[0];
    const int*   spiral[4]; const int* cols[4];
    const float* vals[4];   const float* W[4]; const float* bias[4];
    for (int i = 0; i < 4; ++i) {
        int base = 1 + i * 6;
        spiral[i] = (const int*)  d_in[base + 0];
        // rows{i} at base+1 is implied by structure (repeat(arange, 3)), unused
        cols[i]   = (const int*)  d_in[base + 2];
        vals[i]   = (const float*)d_in[base + 3];
        W[i]      = (const float*)d_in[base + 4];
        bias[i]   = (const float*)d_in[base + 5];
    }
    const float* Wf = (const float*)d_in[25];
    const float* bf = (const float*)d_in[26];

    char* ws = (char*)d_ws;
    float*     y    = (float*)    (ws + OFF_Y);
    float*     h1   = (float*)    (ws + OFF_H1);
    float*     h2   = (float*)    (ws + OFF_H2);
    float*     h3   = (float*)    (ws + OFF_H3);
    float*     h4   = (float*)    (ws + OFF_H4);
    _Float16*  bp0  = (_Float16*) (ws + OFF_BP0);
    _Float16*  bp1  = (_Float16*) (ws + OFF_BP1);
    _Float16*  bp2  = (_Float16*) (ws + OFF_BP2);
    _Float16*  bp3  = (_Float16*) (ws + OFF_BP3);
    _Float16*  bpf  = (_Float16*) (ws + OFF_BPF);
    float*     part = (float*)    (ws + OFF_PART);
    float*     out  = (float*)    d_out;

    const int TB = 256;
    auto cdiv = [](int a, int b) { return (a + b - 1) / b; };

    // ---- pre-pack all weights into B-fragment layout ----
    pack_w_kernel<<<cdiv(1   * 2  * 32, TB), TB, 0, stream>>>(W[0], bp0,    27,  32,   1,  2);
    pack_w_kernel<<<cdiv(9   * 4  * 32, TB), TB, 0, stream>>>(W[1], bp1,   288,  64,   9,  4);
    pack_w_kernel<<<cdiv(18  * 8  * 32, TB), TB, 0, stream>>>(W[2], bp2,   576, 128,  18,  8);
    pack_w_kernel<<<cdiv(36  * 16 * 32, TB), TB, 0, stream>>>(W[3], bp3,  1152, 256,  36, 16);
    pack_w_kernel<<<cdiv(512 * 16 * 32, TB), TB, 0, stream>>>(Wf,   bpf, 16384, 256, 512, 16);

    // ---- block 0: V=16384, C 3->32 ----
    {
        int MT = 64 * 16384 / 16, NT = 2, tiles = MT * NT;        // 131072 waves
        spiral_gemm_kernel<3><<<tiles * 32 / TB, TB, 0, stream>>>(
            x, spiral[0], bp0, bias[0], y, 14, 27, 1, 32, NT, tiles);
        int total = 64 * 4096 * 32;                               // pool -> h1
        pool_kernel<<<cdiv(total, TB), TB, 0, stream>>>(y, cols[0], vals[0], h1, 16384, 5, 12, total);
    }
    // ---- block 1: V=4096, C 32->64 ----
    {
        int MT = 64 * 4096 / 16, NT = 4, tiles = MT * NT;         // 65536 waves
        spiral_gemm_kernel<32><<<tiles * 32 / TB, TB, 0, stream>>>(
            h1, spiral[1], bp1, bias[1], y, 12, 288, 9, 64, NT, tiles);
        int total = 64 * 1024 * 64;                               // pool -> h2
        pool_kernel<<<cdiv(total, TB), TB, 0, stream>>>(y, cols[1], vals[1], h2, 4096, 6, 10, total);
    }
    // ---- block 2: V=1024, C 64->128 ----
    {
        int MT = 64 * 1024 / 16, NT = 8, tiles = MT * NT;         // 32768 waves
        spiral_gemm_kernel<64><<<tiles * 32 / TB, TB, 0, stream>>>(
            h2, spiral[2], bp2, bias[2], y, 10, 576, 18, 128, NT, tiles);
        int total = 64 * 256 * 128;                               // pool -> h3
        pool_kernel<<<cdiv(total, TB), TB, 0, stream>>>(y, cols[2], vals[2], h3, 1024, 7, 8, total);
    }
    // ---- block 3: V=256, C 128->256 ----
    {
        int MT = 64 * 256 / 16, NT = 16, tiles = MT * NT;         // 16384 waves
        spiral_gemm_kernel<128><<<tiles * 32 / TB, TB, 0, stream>>>(
            h3, spiral[3], bp3, bias[3], y, 8, 1152, 36, 256, NT, tiles);
        int total = 64 * 64 * 256;                                // pool -> h4
        pool_kernel<<<cdiv(total, TB), TB, 0, stream>>>(y, cols[3], vals[3], h4, 256, 8, 6, total);
    }
    // ---- final latent: [64,16384] @ [16384,256] + bf ----
    final_gemm_kernel<<<4 * 16 * 16 * 32 / TB, TB, 0, stream>>>(h4, bpf, part);
    final_reduce_kernel<<<cdiv(16384, TB), TB, 0, stream>>>(part, bf, out);
}
